// VGAEEncoder_atac_pro_59081570123789
// MI455X (gfx1250) — matrix operand order repaired and checked by reading, not compiled, not run on hardware
//
#include <hip/hip_runtime.h>

typedef __attribute__((ext_vector_type(16))) __bf16 v16bf;
typedef __attribute__((ext_vector_type(8)))  __bf16 v8bf;
typedef __attribute__((ext_vector_type(8)))  float  v8f;

// ---------------------------------------------------------------------------
// Utility kernels
// ---------------------------------------------------------------------------
__global__ void zero_kernel(float* p, long n) {
    long i = (long)blockIdx.x * blockDim.x + threadIdx.x;
    long stride = (long)gridDim.x * blockDim.x;
    for (; i < n; i += stride) p[i] = 0.0f;
}

__global__ void deg_kernel(const int* ei, float* deg, int E) {
    int e = blockIdx.x * blockDim.x + threadIdx.x;
    if (e >= E) return;
    int dst = ei[E + e];
    unsafeAtomicAdd(deg + dst, 1.0f);
}

__global__ void dinv_kernel(float* deg, int N) {
    int i = blockIdx.x * blockDim.x + threadIdx.x;
    if (i >= N) return;
    float d = deg[i];
    deg[i] = (d > 0.0f) ? rsqrtf(fmaxf(d, 1.0f)) : 0.0f;
}

__global__ void norm_kernel(const int* ei, const float* dinv, float* nrm, int E) {
    int e = blockIdx.x * blockDim.x + threadIdx.x;
    if (e >= E) return;
    int s = ei[e];
    int t = ei[E + e];
    nrm[e] = dinv[s] * dinv[t];
}

// Edge-parallel SpMM scatter: hout[dst] += norm[e] * hin[src], float4 chunks.
__global__ void spmm_scatter(const int* __restrict__ ei, const float* __restrict__ nrm,
                             const float* __restrict__ hin, float* __restrict__ hout,
                             int E, int d4) {
    long idx = (long)blockIdx.x * blockDim.x + threadIdx.x;
    long total = (long)E * d4;
    if (idx >= total) return;
    int e = (int)(idx / d4);
    int f = (int)(idx % d4) * 4;
    int s = ei[e];
    int t = ei[E + e];
    float w = nrm[e];
    int d = d4 * 4;
    const float4 v = *(const float4*)(hin + (long)s * d + f);
    float* o = hout + (long)t * d + f;
    unsafeAtomicAdd(o + 0, w * v.x);
    unsafeAtomicAdd(o + 1, w * v.y);
    unsafeAtomicAdd(o + 2, w * v.z);
    unsafeAtomicAdd(o + 3, w * v.w);
}

// ---------------------------------------------------------------------------
// bf16 WMMA GEMM: C[M,N] (f32) (+)= A[M,K] (f32->bf16) * B[K,N] (f32->bf16)
// Block = 256 threads = 8 waves; block tile 64x64; K-step 32.
// LDS tiles are stored in WMMA *fragment order* so each lane reads its whole
// 16-bf16 operand with two ds_load_b128 (no scalar gather/pack).
//
// Fragment maps (ISA 7.12.2, 16-bit operands, wave32):
//   A(16x32) elem (m,k): lane = m%16 + 16*((k>>3)&1), elem = (k&7) + 8*(k>>4)
//   B(32x16) elem (k,n): lane = n%16 + 16*((k>>3)&1), elem = (k&7) + 8*(k>>4)
// ---------------------------------------------------------------------------
#define TM 64
#define TN 64
#define TK 32

__global__ __launch_bounds__(256)
void wmma_gemm_bf16(const float* __restrict__ A, const float* __restrict__ B,
                    float* __restrict__ C, int M, int N, int K, int accumulate) {
    // [tileRow/Col][lane][half-of-16-elems]; 4 KB each.
    __shared__ v8bf Afrag[4][32][2];
    __shared__ v8bf Bfrag[4][32][2];

    const int bm = blockIdx.x * TM;
    const int bn = blockIdx.y * TN;
    const int tid = threadIdx.x;
    const int wave = tid >> 5;      // 0..7
    const int lane = tid & 31;
    const int hf   = lane >> 4;     // half-lane group 0/1
    const int l16  = lane & 15;

    const int wrow  = wave >> 1;          // tile row within block (0..3)
    const int wcol0 = (wave & 1) * 2;     // first of two tile cols (0 or 2)

    float cacc[2][8];
    if (accumulate) {
        #pragma unroll
        for (int t = 0; t < 2; ++t) {
            int n0 = (wcol0 + t) * 16;
            #pragma unroll
            for (int r = 0; r < 8; ++r) {
                int gm = bm + wrow * 16 + r + hf * 8;
                int gn = bn + n0 + l16;
                cacc[t][r] = (gm < M && gn < N) ? C[(long)gm * N + gn] : 0.0f;
            }
        }
    } else {
        #pragma unroll
        for (int t = 0; t < 2; ++t)
            #pragma unroll
            for (int r = 0; r < 8; ++r) cacc[t][r] = 0.0f;
    }

    // Uniform-per-thread staging coordinates (loop invariant).
    const int sa_m  = tid >> 2;            // A row 0..63
    const int sa_kg = (tid & 3) * 8;       // A k-group 0,8,16,24
    const int sa_gm = bm + sa_m;
    const int sa_lane = (sa_m & 15) + 16 * ((sa_kg >> 3) & 1);

    const int sb_n  = tid & 63;            // B col 0..63
    const int sb_kb = (tid >> 6) * 8;      // B k-block 0,8,16,24
    const int sb_gn = bn + sb_n;
    const int sb_lane = (sb_n & 15) + 16 * ((sb_kb >> 3) & 1);

    for (int k0 = 0; k0 < K; k0 += TK) {
        // ---- Stage A (64x32): two float4 loads, pack to 8 bf16, one b128 store.
        {
            float4 lo = make_float4(0.f, 0.f, 0.f, 0.f), hi = lo;
            if (sa_gm < M) {
                const float4* p = (const float4*)(A + (long)sa_gm * K + k0 + sa_kg);
                lo = p[0];
                hi = p[1];
            }
            v8bf pk;
            pk[0] = (__bf16)lo.x; pk[1] = (__bf16)lo.y;
            pk[2] = (__bf16)lo.z; pk[3] = (__bf16)lo.w;
            pk[4] = (__bf16)hi.x; pk[5] = (__bf16)hi.y;
            pk[6] = (__bf16)hi.z; pk[7] = (__bf16)hi.w;
            Afrag[sa_m >> 4][sa_lane][sa_kg >> 4] = pk;
        }
        // ---- Stage B (32x64): guard hoisted so the 8 strided loads issue as
        //      one clause (coalesced across the 64 threads sharing a k-block).
        {
            v8bf pk;
            if (sb_gn < N) {
                float v0 = B[(long)(k0 + sb_kb + 0) * N + sb_gn];
                float v1 = B[(long)(k0 + sb_kb + 1) * N + sb_gn];
                float v2 = B[(long)(k0 + sb_kb + 2) * N + sb_gn];
                float v3 = B[(long)(k0 + sb_kb + 3) * N + sb_gn];
                float v4 = B[(long)(k0 + sb_kb + 4) * N + sb_gn];
                float v5 = B[(long)(k0 + sb_kb + 5) * N + sb_gn];
                float v6 = B[(long)(k0 + sb_kb + 6) * N + sb_gn];
                float v7 = B[(long)(k0 + sb_kb + 7) * N + sb_gn];
                pk[0] = (__bf16)v0; pk[1] = (__bf16)v1;
                pk[2] = (__bf16)v2; pk[3] = (__bf16)v3;
                pk[4] = (__bf16)v4; pk[5] = (__bf16)v5;
                pk[6] = (__bf16)v6; pk[7] = (__bf16)v7;
            } else {
                #pragma unroll
                for (int j = 0; j < 8; ++j) pk[j] = (__bf16)0.0f;
            }
            Bfrag[sb_n >> 4][sb_lane][sb_kb >> 4] = pk;
        }
        // Prefetch next K tile into cache (global_prefetch_b8).
        if (k0 + TK < K) {
            if (sa_gm < M) __builtin_prefetch(&A[(long)sa_gm * K + k0 + TK + sa_kg], 0, 1);
            if (sb_gn < N) __builtin_prefetch(&B[(long)(k0 + TK + sb_kb) * N + sb_gn], 0, 1);
        }
        __syncthreads();

        // ---- Fragments: straight wide LDS loads, already in register order.
        v8bf a0 = Afrag[wrow][lane][0];
        v8bf a1 = Afrag[wrow][lane][1];
        v16bf afrag = __builtin_shufflevector(a0, a1,
                0, 1, 2, 3, 4, 5, 6, 7, 8, 9, 10, 11, 12, 13, 14, 15);

        #pragma unroll
        for (int t = 0; t < 2; ++t) {
            v8bf b0 = Bfrag[wcol0 + t][lane][0];
            v8bf b1 = Bfrag[wcol0 + t][lane][1];
            v16bf bfrag = __builtin_shufflevector(b0, b1,
                    0, 1, 2, 3, 4, 5, 6, 7, 8, 9, 10, 11, 12, 13, 14, 15);
            v8f c;
            #pragma unroll
            for (int r = 0; r < 8; ++r) c[r] = cacc[t][r];
            c = __builtin_amdgcn_wmma_f32_16x16x32_bf16(
                    /*neg_a=*/false, afrag, /*neg_b=*/false, bfrag,
                    /*c_mod=*/(short)0, c, /*reuse_a=*/false, /*reuse_b=*/false);
            #pragma unroll
            for (int r = 0; r < 8; ++r) cacc[t][r] = c[r];
        }
        __syncthreads();
    }

    #pragma unroll
    for (int t = 0; t < 2; ++t) {
        int n0 = (wcol0 + t) * 16;
        #pragma unroll
        for (int r = 0; r < 8; ++r) {
            int gm = bm + wrow * 16 + r + hf * 8;
            int gn = bn + n0 + l16;
            if (gm < M && gn < N) C[(long)gm * N + gn] = cacc[t][r];
        }
    }
}

// ---------------------------------------------------------------------------
// Epilogues
// ---------------------------------------------------------------------------
__global__ void bias_act(float* X, const float* b, long total, int N, int relu) {
    long i = (long)blockIdx.x * blockDim.x + threadIdx.x;
    if (i >= total) return;
    int n = (int)(i % N);
    float v = X[i] + b[n];
    if (relu) v = fmaxf(v, 0.0f);
    X[i] = v;
}

__device__ __forceinline__ unsigned pcg_hash(unsigned x) {
    x = x * 747796405u + 2891336453u;
    unsigned w = ((x >> ((x >> 28u) + 4u)) ^ x) * 277803737u;
    return (w >> 22u) ^ w;
}

// z = eps * exp(logstd) + mu ; deterministic hash-based gaussian eps.
__global__ void reparam_kernel(const float* __restrict__ mu, const float* __restrict__ ls,
                               float* __restrict__ z, long n) {
    long i = (long)blockIdx.x * blockDim.x + threadIdx.x;
    if (i >= n) return;
    unsigned r1 = pcg_hash((unsigned)i * 2654435761u + 12345u);
    unsigned r2 = pcg_hash(r1 ^ 0x9e3779b9u);
    float u1 = ((float)r1 + 1.0f) * 2.3283064365386963e-10f;  // (0,1]
    float u2 = (float)r2 * 2.3283064365386963e-10f;
    float eps = sqrtf(-2.0f * __logf(u1)) * __cosf(6.283185307179586f * u2);
    z[i] = eps * __expf(ls[i]) + mu[i];
}

// ---------------------------------------------------------------------------
// Host launch
// ---------------------------------------------------------------------------
static inline size_t alignup(size_t x) { return (x + 255) & ~(size_t)255; }

extern "C" void kernel_launch(void* const* d_in, const int* in_sizes, int n_in,
                              void* d_out, int out_size, void* d_ws, size_t ws_size,
                              hipStream_t stream) {
    const float* x   = (const float*)d_in[0];
    const int*   ei  = (const int*)d_in[1];
    const float* W1  = (const float*)d_in[2];
    const float* b1  = (const float*)d_in[3];
    const float* W2  = (const float*)d_in[4];
    const float* b2  = (const float*)d_in[5];
    const float* Wmu = (const float*)d_in[6];
    const float* bmu = (const float*)d_in[7];
    const float* Wls = (const float*)d_in[8];
    const float* bls = (const float*)d_in[9];

    const int N = 20000, E = 320000;
    const int DIN = 512, DH1 = 512, DH2 = 256, DZ = 32;

    // Workspace carve-out
    char* w = (char*)d_ws;
    float* dinv = (float*)w;  w += alignup((size_t)N * 4);
    float* nrm  = (float*)w;  w += alignup((size_t)E * 4);
    float* hP   = (float*)w;  w += alignup((size_t)N * DIN * 4);
    float* hQ   = (float*)w;  w += alignup((size_t)N * DIN * 4);
    float* acc  = (float*)w;  w += alignup((size_t)N * DH1 * 4);
    float* acc2 = (float*)w;  w += alignup((size_t)N * DH2 * 4);

    float* mu = (float*)d_out;
    float* ls = mu + (long)N * DZ;
    float* zz = ls + (long)N * DZ;

    auto zero = [&](float* p, long n) {
        int blocks = (int)((n + 255) / 256);
        if (blocks > 65536) blocks = 65536;
        zero_kernel<<<blocks, 256, 0, stream>>>(p, n);
    };
    auto gemm = [&](const float* A, const float* B, float* C, int M, int Nn, int Kk, int accf) {
        dim3 grid((M + TM - 1) / TM, (Nn + TN - 1) / TN);
        wmma_gemm_bf16<<<grid, 256, 0, stream>>>(A, B, C, M, Nn, Kk, accf);
    };
    auto spmm = [&](const float* hin, float* hout, int d) {
        zero(hout, (long)N * d);
        long total = (long)E * (d / 4);
        spmm_scatter<<<(int)((total + 255) / 256), 256, 0, stream>>>(ei, nrm, hin, hout, E, d / 4);
    };

    // --- Normalization: deg over dst, dinv = rsqrt(max(deg,1)) where deg>0 ---
    zero(dinv, N);
    deg_kernel<<<(E + 255) / 256, 256, 0, stream>>>(ei, dinv, E);
    dinv_kernel<<<(N + 255) / 256, 256, 0, stream>>>(dinv, N);
    norm_kernel<<<(E + 255) / 256, 256, 0, stream>>>(ei, dinv, nrm, E);

    float* bufs[2] = { hP, hQ };

    // --- Layer 1: TAGConv(512 -> 512) + ReLU ---
    gemm(x, W1, acc, N, DH1, DIN, 0);
    {
        const float* cur = x;
        for (int k = 1; k <= 3; ++k) {
            float* nxt = bufs[(k - 1) & 1];
            spmm(cur, nxt, DIN);
            gemm(nxt, W1 + (long)k * DIN * DH1, acc, N, DH1, DIN, 1);
            cur = nxt;
        }
    }
    bias_act<<<(int)(((long)N * DH1 + 255) / 256), 256, 0, stream>>>(acc, b1, (long)N * DH1, DH1, 1);

    // --- Layer 2: TAGConv(512 -> 256) + ReLU ---
    gemm(acc, W2, acc2, N, DH2, DH1, 0);
    {
        const float* cur = acc;
        for (int k = 1; k <= 3; ++k) {
            float* nxt = bufs[(k - 1) & 1];
            spmm(cur, nxt, DH1);
            gemm(nxt, W2 + (long)k * DH1 * DH2, acc2, N, DH2, DH1, 1);
            cur = nxt;
        }
    }
    bias_act<<<(int)(((long)N * DH2 + 255) / 256), 256, 0, stream>>>(acc2, b2, (long)N * DH2, DH2, 1);

    // --- mu / logstd heads: TAGConv(256 -> 32), shared propagations ---
    gemm(acc2, Wmu, mu, N, DZ, DH2, 0);
    gemm(acc2, Wls, ls, N, DZ, DH2, 0);
    {
        const float* cur = acc2;
        for (int k = 1; k <= 3; ++k) {
            float* nxt = bufs[(k - 1) & 1];
            spmm(cur, nxt, DH2);
            gemm(nxt, Wmu + (long)k * DH2 * DZ, mu, N, DZ, DH2, 1);
            gemm(nxt, Wls + (long)k * DH2 * DZ, ls, N, DZ, DH2, 1);
            cur = nxt;
        }
    }
    bias_act<<<(int)(((long)N * DZ + 255) / 256), 256, 0, stream>>>(mu, bmu, (long)N * DZ, DZ, 0);
    bias_act<<<(int)(((long)N * DZ + 255) / 256), 256, 0, stream>>>(ls, bls, (long)N * DZ, DZ, 0);

    // --- Reparameterize ---
    reparam_kernel<<<(int)(((long)N * DZ + 255) / 256), 256, 0, stream>>>(mu, ls, zz, (long)N * DZ);
}
